// BidirectionalCrossAttention_6236292514541
// MI455X (gfx1250) — matrix operand order repaired and checked by reading, not compiled, not run on hardware
//
#include <hip/hip_runtime.h>

typedef _Float16 half8  __attribute__((ext_vector_type(8)));
typedef _Float16 half16 __attribute__((ext_vector_type(16)));
typedef float    float8 __attribute__((ext_vector_type(8)));
typedef float    float4_t __attribute__((ext_vector_type(4)));
typedef unsigned int u32x4 __attribute__((ext_vector_type(4)));
typedef int          i32x4 __attribute__((ext_vector_type(4)));
typedef int          i32x8 __attribute__((ext_vector_type(8)));

// LDS row pitch (in halves) for 32-wide K tiles: 32 + 8 pad keeps 16B alignment
// for all fragment sub-loads and staggers banks. TDM reproduces this pitch via
// pad_enable (pad_interval=3 -> every 16 DWORDs, pad_amount=3 -> +4 DWORDs).
#define LDT 40

// ---------------- TDM: 2D f16 tile (32 wide x rows) global -> LDS -----------
__device__ __forceinline__ unsigned lds_off(const void* p) {
  return (unsigned)(unsigned long)p;   // generic LDS pointer: low 32 bits = LDS byte offset
}

__device__ __forceinline__ void tdm_load_tile_f16(const _Float16* gsrc,
                                                  unsigned ldsaddr,
                                                  unsigned rows,
                                                  long row_stride_elems) {
  unsigned long ga = (unsigned long)(const void*)gsrc;
  u32x4 g0;
  g0[0] = 1u;                                            // count=1, no gather
  g0[1] = ldsaddr;                                       // lds_addr
  g0[2] = (unsigned)ga;                                  // global_addr[31:0]
  g0[3] = (unsigned)((ga >> 32) & 0x01FFFFFFu)           // global_addr[56:32]
          | (2u << 30);                                  // type=2 (image)
  i32x8 g1;
  g1[0] = (int)((1u << 16)                               // data_size = 2 bytes
                | (1u << 20)                             // pad_enable
                | (3u << 22)                             // pad_interval: 16 DWORDs
                | (3u << 25));                           // pad_amount: 4 DWORDs
  g1[1] = (int)(32u << 16);                              // tensor_dim0 = 32 (lo16 -> [31:16])
  g1[2] = (int)(rows << 16);                             // tensor_dim0 hi=0 | tensor_dim1 lo
  g1[3] = (int)(32u << 16);                              // tensor_dim1 hi=0 | tile_dim0 = 32
  g1[4] = (int)rows;                                     // tile_dim1 = rows, tile_dim2 = 0
  g1[5] = (int)(unsigned)(row_stride_elems & 0xFFFFFFFFL);   // dim0_stride lo
  g1[6] = (int)(unsigned)((row_stride_elems >> 32) & 0xFFFFL); // dim0_stride hi, dim1_stride lo=0
  g1[7] = 0;
  i32x4 z4 = {0, 0, 0, 0};
  i32x8 z8 = {0, 0, 0, 0, 0, 0, 0, 0};
  __builtin_amdgcn_tensor_load_to_lds(g0, g1, z4, z4, z8, 0);
}

// ---------------- WMMA fragment helpers (per ISA 7.12.2 layouts) ------------
__device__ __forceinline__ half16 frag_a(const _Float16* t) {
  int lane = threadIdx.x & 31;
  int row  = lane & 15;
  int kb   = (lane >> 4) << 3;                 // 0 or 8
  const half8* p0 = (const half8*)(t + row * LDT + kb);
  const half8* p1 = (const half8*)(t + row * LDT + 16 + kb);
  half8 lo = *p0, hi = *p1;
  return __builtin_shufflevector(lo, hi, 0,1,2,3,4,5,6,7,8,9,10,11,12,13,14,15);
}

__device__ __forceinline__ half16 frag_b(const _Float16* t) {
  int lane = threadIdx.x & 31;
  int col  = lane & 15;
  int kb   = (lane >> 4) << 3;                 // 0 or 8
  const half8* p0 = (const half8*)(t + col * LDT + kb);
  const half8* p1 = (const half8*)(t + col * LDT + 16 + kb);
  half8 lo = *p0, hi = *p1;
  return __builtin_shufflevector(lo, hi, 0,1,2,3,4,5,6,7,8,9,10,11,12,13,14,15);
}

// ---------------- Kernel 1: fp32 projection GEMM -> f16 ---------------------
__global__ __launch_bounds__(256)
void proj_kernel(const float* __restrict__ X, const float* __restrict__ W,
                 const float* __restrict__ bias, _Float16* __restrict__ Out,
                 int N, int Kd, float scale) {
  __shared__ __align__(16) _Float16 As[2][128 * LDT];
  __shared__ __align__(16) _Float16 Bs[2][128 * LDT];
  const int m0 = blockIdx.x * 128;
  const int n0 = blockIdx.y * 128;
  const int t = threadIdx.x, wave = t >> 5, lane = t & 31;

  float8 acc[8] = {};
  float4_t va[4];
  float    vb[16];

  auto load_global = [&](int k0) {
#pragma unroll
    for (int i = 0; i < 4; ++i) {
      int idx = t + 256 * i;
      int row = idx >> 3, c4 = idx & 7;
      const float* p = X + (long)(m0 + row) * Kd + k0 + 4 * c4;
      va[i] = *(const float4_t*)p;
      if (k0 + 64 < Kd) __builtin_prefetch(p + 64, 0, 1);
    }
#pragma unroll
    for (int i = 0; i < 16; ++i) {
      int idx = t + 256 * i;
      int nn = idx & 127, kk = idx >> 7;
      const float* p = W + (long)(k0 + kk) * N + n0 + nn;
      vb[i] = *p;
      if (i == 0 && k0 + 64 < Kd) __builtin_prefetch(p + 64L * N, 0, 1);
    }
  };
  auto store_lds = [&](int buf) {
#pragma unroll
    for (int i = 0; i < 4; ++i) {
      int idx = t + 256 * i;
      int row = idx >> 3, c4 = idx & 7;
      _Float16* d = &As[buf][row * LDT + 4 * c4];
      d[0] = (_Float16)va[i][0]; d[1] = (_Float16)va[i][1];
      d[2] = (_Float16)va[i][2]; d[3] = (_Float16)va[i][3];
    }
#pragma unroll
    for (int i = 0; i < 16; ++i) {
      int idx = t + 256 * i;
      int nn = idx & 127, kk = idx >> 7;
      Bs[buf][nn * LDT + kk] = (_Float16)vb[i];
    }
  };

  load_global(0);
  const int nk = Kd / 32;
  for (int ks = 0; ks < nk; ++ks) {
    const int buf = ks & 1;
    store_lds(buf);
    if (ks + 1 < nk) load_global(32 * (ks + 1));
    __syncthreads();
    half16 a = frag_a(&As[buf][16 * wave * LDT]);
    half16 bf[8];
#pragma unroll
    for (int j = 0; j < 8; ++j) bf[j] = frag_b(&Bs[buf][16 * j * LDT]);
#pragma unroll
    for (int j = 0; j < 8; ++j)
      acc[j] = __builtin_amdgcn_wmma_f32_16x16x32_f16(
          false, a, false, bf[j], (short)0, acc[j], false, false);
  }

  const int col = lane & 15, rh = lane >> 4;
#pragma unroll
  for (int j = 0; j < 8; ++j) {
    int n = n0 + 16 * j + col;
    float bv = bias[n];
#pragma unroll
    for (int r = 0; r < 8; ++r) {
      int m = m0 + 16 * wave + r + 8 * rh;
      Out[(long)m * N + n] = (_Float16)((acc[j][r] + bv) * scale);
    }
  }
}

// ---------------- Kernel 2: scores = Q . K^T — pure TDM + WMMA --------------
__global__ __launch_bounds__(256)
void scores_kernel(const _Float16* __restrict__ Qh, const _Float16* __restrict__ Kh,
                   _Float16* __restrict__ P, int S, int Dh) {
  __shared__ __align__(16) _Float16 As[2][128 * LDT];
  __shared__ __align__(16) _Float16 Bs[2][128 * LDT];
  const _Float16* A = Qh + (long)blockIdx.z * S * Dh;
  const _Float16* B = Kh + (long)blockIdx.z * S * Dh;
  _Float16* C = P + (long)blockIdx.z * S * S;
  const int m0 = blockIdx.x * 128;
  const int n0 = blockIdx.y * 128;
  const int t = threadIdx.x, wave = t >> 5, lane = t & 31;
  const bool issuer = (wave == 0);

  float8 acc[8] = {};

  if (issuer) {
    tdm_load_tile_f16(A + (long)m0 * Dh, lds_off(&As[0][0]), 128, Dh);
    tdm_load_tile_f16(B + (long)n0 * Dh, lds_off(&Bs[0][0]), 128, Dh);
  }

  const int nk = Dh / 32;
  for (int ks = 0; ks < nk; ++ks) {
    const int buf = ks & 1;
    if (issuer) __builtin_amdgcn_s_wait_tensorcnt(0);  // tile ks landed in LDS
    __syncthreads();                                    // everyone done with buf^1
    if (ks + 1 < nk && issuer) {                        // DMA next tile into buf^1
      tdm_load_tile_f16(A + (long)m0 * Dh + 32 * (ks + 1),
                        lds_off(&As[buf ^ 1][0]), 128, Dh);
      tdm_load_tile_f16(B + (long)n0 * Dh + 32 * (ks + 1),
                        lds_off(&Bs[buf ^ 1][0]), 128, Dh);
    }
    half16 a = frag_a(&As[buf][16 * wave * LDT]);
    half16 bf[8];
#pragma unroll
    for (int j = 0; j < 8; ++j) bf[j] = frag_b(&Bs[buf][16 * j * LDT]);
#pragma unroll
    for (int j = 0; j < 8; ++j)
      acc[j] = __builtin_amdgcn_wmma_f32_16x16x32_f16(
          false, a, false, bf[j], (short)0, acc[j], false, false);
  }

  const int col = lane & 15, rh = lane >> 4;
#pragma unroll
  for (int j = 0; j < 8; ++j) {
    int n = n0 + 16 * j + col;
#pragma unroll
    for (int r = 0; r < 8; ++r) {
      int m = m0 + 16 * wave + r + 8 * rh;
      C[(long)m * S + n] = (_Float16)acc[j][r];
    }
  }
}

// ---------------- Kernel 3: row softmax, in place (f16) ---------------------
__global__ __launch_bounds__(256)
void softmax_kernel(_Float16* __restrict__ P, int S) {
  _Float16* p = P + (long)blockIdx.x * S;
  __shared__ float red[256];
  const int t = threadIdx.x;
  float x[8];
  float m = -3.0e38f;
#pragma unroll
  for (int i = 0; i < 8; ++i) { x[i] = (float)p[t + 256 * i]; m = fmaxf(m, x[i]); }
  red[t] = m; __syncthreads();
  for (int s = 128; s > 0; s >>= 1) {
    if (t < s) red[t] = fmaxf(red[t], red[t + s]);
    __syncthreads();
  }
  m = red[0]; __syncthreads();
  float sum = 0.0f;
#pragma unroll
  for (int i = 0; i < 8; ++i) { x[i] = __expf(x[i] - m); sum += x[i]; }
  red[t] = sum; __syncthreads();
  for (int s = 128; s > 0; s >>= 1) {
    if (t < s) red[t] += red[t + s];
    __syncthreads();
  }
  float inv = 1.0f / red[0];
#pragma unroll
  for (int i = 0; i < 8; ++i) p[t + 256 * i] = (_Float16)(x[i] * inv);
}

// ---------------- Kernel 4: out = P . V (A via TDM, V staged/transposed) ----
__global__ __launch_bounds__(256)
void out_kernel(const _Float16* __restrict__ P, const _Float16* __restrict__ Vh,
                float* __restrict__ O, int S, int Dv) {
  __shared__ __align__(16) _Float16 As[2][128 * LDT];
  __shared__ __align__(16) _Float16 Bs[2][128 * LDT];
  const _Float16* A = P + (long)blockIdx.z * S * S;
  const _Float16* B = Vh + (long)blockIdx.z * S * Dv;
  float* C = O + (long)blockIdx.z * S * Dv;
  const int m0 = blockIdx.x * 128;
  const int n0 = blockIdx.y * 128;
  const int t = threadIdx.x, wave = t >> 5, lane = t & 31;
  const bool issuer = (wave == 0);

  float8 acc[8] = {};
  _Float16 vb[16];

  auto load_b = [&](int k0) {
#pragma unroll
    for (int i = 0; i < 16; ++i) {             // V tile 32x128 f16
      int idx = t + 256 * i;
      int nn = idx & 127, kk = idx >> 7;
      const _Float16* pb = B + (long)(k0 + kk) * Dv + n0 + nn;
      vb[i] = *pb;
      if (i == 0 && k0 + 64 < S) __builtin_prefetch(pb + 64L * Dv, 0, 1);
    }
  };
  auto store_b = [&](int buf) {
#pragma unroll
    for (int i = 0; i < 16; ++i) {
      int idx = t + 256 * i;
      int nn = idx & 127, kk = idx >> 7;
      Bs[buf][nn * LDT + kk] = vb[i];
    }
  };

  load_b(0);
  if (issuer) tdm_load_tile_f16(A + (long)m0 * S, lds_off(&As[0][0]), 128, S);

  const int nk = S / 32;
  for (int ks = 0; ks < nk; ++ks) {
    const int buf = ks & 1;
    store_b(buf);
    if (issuer) __builtin_amdgcn_s_wait_tensorcnt(0);
    __syncthreads();
    if (ks + 1 < nk) {
      load_b(32 * (ks + 1));
      if (issuer)
        tdm_load_tile_f16(A + (long)m0 * S + 32 * (ks + 1),
                          lds_off(&As[buf ^ 1][0]), 128, S);
    }
    half16 a = frag_a(&As[buf][16 * wave * LDT]);
    half16 bf[8];
#pragma unroll
    for (int j = 0; j < 8; ++j) bf[j] = frag_b(&Bs[buf][16 * j * LDT]);
#pragma unroll
    for (int j = 0; j < 8; ++j)
      acc[j] = __builtin_amdgcn_wmma_f32_16x16x32_f16(
          false, a, false, bf[j], (short)0, acc[j], false, false);
  }

  const int col = lane & 15, rh = lane >> 4;
#pragma unroll
  for (int j = 0; j < 8; ++j) {
    int n = n0 + 16 * j + col;
#pragma unroll
    for (int r = 0; r < 8; ++r) {
      int m = m0 + 16 * wave + r + 8 * rh;
      C[(long)m * Dv + n] = acc[j][r];
    }
  }
}

// ---------------- Host launcher ---------------------------------------------
extern "C" void kernel_launch(void* const* d_in, const int* in_sizes, int n_in,
                              void* d_out, int out_size, void* d_ws, size_t ws_size,
                              hipStream_t stream) {
  constexpr long B = 8, S = 2048, D = 1024;
  const float* q  = (const float*)d_in[0];
  const float* k  = (const float*)d_in[1];
  const float* v  = (const float*)d_in[2];
  const float* Wq = (const float*)d_in[3];
  const float* bq = (const float*)d_in[4];
  const float* Wk = (const float*)d_in[5];
  const float* bk = (const float*)d_in[6];
  const float* Wv = (const float*)d_in[7];
  const float* bv = (const float*)d_in[8];
  float* out = (float*)d_out;

  _Float16* Qh = (_Float16*)d_ws;
  _Float16* Kh = Qh + B * S * D;
  _Float16* Vh = Kh + B * S * D;
  _Float16* P  = Vh + B * S * D;     // B*S*S halves

  dim3 blk(256);
  const float scale = 1.0f / 32.0f;  // 1/sqrt(1024), folded into Q

  dim3 gproj((B * S) / 128, D / 128);
  proj_kernel<<<gproj, blk, 0, stream>>>(q, Wq, bq, Qh, (int)D, (int)D, scale);
  proj_kernel<<<gproj, blk, 0, stream>>>(k, Wk, bk, Kh, (int)D, (int)D, 1.0f);
  proj_kernel<<<gproj, blk, 0, stream>>>(v, Wv, bv, Vh, (int)D, (int)D, 1.0f);

  dim3 gs(S / 128, S / 128, B);
  scores_kernel<<<gs, blk, 0, stream>>>(Qh, Kh, P, (int)S, (int)D);

  softmax_kernel<<<dim3(B * S), blk, 0, stream>>>(P, (int)S);

  dim3 go(S / 128, D / 128, B);
  out_kernel<<<go, blk, 0, stream>>>(P, Vh, out, (int)S, (int)D);
}
